// adapter_46059229282675
// MI455X (gfx1250) — matrix operand order repaired and checked by reading, not compiled, not run on hardware
//
#include <hip/hip_runtime.h>

typedef _Float16 v16h __attribute__((ext_vector_type(16)));
typedef _Float16 v8h  __attribute__((ext_vector_type(8)));
typedef _Float16 v4h  __attribute__((ext_vector_type(4)));
typedef _Float16 v2h  __attribute__((ext_vector_type(2)));
typedef float    v8f  __attribute__((ext_vector_type(8)));

#define CDIM 768
#define HDIM 48
#define WAVES 4
#define Y_STRIDE 776u   // f16 elems per y row (768 + 8 pad -> row stride = 4 banks mod 64)
#define HID_STRIDE 72u  // f16 elems per hid row (64 + 8 pad)

// Weight fragment pool: W1 = 72 frags, W2 = 96 frags, 1024 B each, contiguous.
#define W1F_BYTES 73728u
#define W2F_BYTES 98304u
#define WF_BYTES  172032u   // W1F_BYTES + W2F_BYTES

// LDS byte offsets (dynamic shared memory)
#define OFF_W1F   0u
#define OFF_W2F   73728u
#define OFF_GAMMA 172032u   // 768 f32
#define OFF_BETA  175104u   // 768 f32
#define OFF_B1    178176u   // 48 f32
#define OFF_B2    178368u   // 768 f32
#define OFF_Y     181440u   // per-wave: 16 * 776 * 2 = 24832 B
#define Y_BYTES   24832u
#define OFF_HID   280768u   // per-wave: 16 * 72 * 2 = 2304 B
#define HID_BYTES 2304u
#define LDS_TOTAL 289984u

static __device__ inline v16h cat8(v8h lo, v8h hi) {
  return __builtin_shufflevector(lo, hi, 0,1,2,3,4,5,6,7,8,9,10,11,12,13,14,15);
}

// ---- shared fragment-swizzle helpers (used by prep kernel and fallback path) ----
// B k order used here (contiguous, same as A): lanes 0-15 hold K=0..15 (V_j: K=2j,2j+1), lanes 16-31 hold K=16..31.
static __device__ inline void stage_one_w1(int u, const float* __restrict__ W1, char* base) {
  int f = u >> 8, rem = u & 255;
  int l = rem >> 3, j = rem & 7;
  int kc = f / 3, nt = f - kc * 3;        // frag = kc*3 + nt
  int K = kc * 32 + ((l >> 4) << 4) + (j << 1);
  int N = nt * 16 + (l & 15);
  v2h p; p[0] = (_Float16)W1[K * HDIM + N]; p[1] = (_Float16)W1[(K + 1) * HDIM + N];
  *(v2h*)(base + (unsigned)f * 1024u + (unsigned)(j >> 2) * 512u
          + (unsigned)l * 16u + (unsigned)(j & 3) * 4u) = p;
}
static __device__ inline void stage_one_w2(int u, const float* __restrict__ W2, char* base) {
  int f = u >> 8, rem = u & 255;
  int l = rem >> 3, j = rem & 7;
  int nt = f >> 1, kc = f & 1;            // frag = nt*2 + kc, K padded 48->64
  int K = kc * 32 + ((l >> 4) << 4) + (j << 1);
  int N = nt * 16 + (l & 15);
  v2h p;
  p[0] = (K     < HDIM) ? (_Float16)W2[K       * CDIM + N] : (_Float16)0.f;
  p[1] = (K + 1 < HDIM) ? (_Float16)W2[(K + 1) * CDIM + N] : (_Float16)0.f;
  *(v2h*)(base + (unsigned)f * 1024u + (unsigned)(j >> 2) * 512u
          + (unsigned)l * 16u + (unsigned)(j & 3) * 4u) = p;
}

// ---- one-shot weight conversion fp32 -> f16 fragment pool in d_ws ----
__global__ __launch_bounds__(256, 4)
void prep_weights_kernel(const float* __restrict__ W1, const float* __restrict__ W2,
                         char* __restrict__ ws) {
  int u = blockIdx.x * 256 + threadIdx.x;           // [0, 43008)
  if (u < 72 * 256)       stage_one_w1(u, W1, ws);
  else                    stage_one_w2(u - 72 * 256, W2, ws + OFF_W2F);
}

__global__ __launch_bounds__(128, 1)
void fused_ln_mlp_kernel(const float* __restrict__ x,
                         const float* __restrict__ W1,
                         const float* __restrict__ b1,
                         const float* __restrict__ W2,
                         const float* __restrict__ b2,
                         const float* __restrict__ gamma,
                         const float* __restrict__ beta,
                         float* __restrict__ out,
                         const char* __restrict__ wsrc,  // pre-swizzled f16 fragments
                         int use_ws)
{
  extern __shared__ char smem[];
  const int tid  = threadIdx.x;
  const int lane = tid & 31;
  const int wave = tid >> 5;

  float* gam = (float*)(smem + OFF_GAMMA);
  float* bet = (float*)(smem + OFF_BETA);
  float* b1l = (float*)(smem + OFF_B1);
  float* b2l = (float*)(smem + OFF_B2);
  _Float16* ylds = (_Float16*)(smem + OFF_Y   + (unsigned)wave * Y_BYTES);
  _Float16* hlds = (_Float16*)(smem + OFF_HID + (unsigned)wave * HID_BYTES);

  // ---- stage per-channel vectors into LDS ----
  for (int i = tid; i < CDIM; i += 128) { gam[i] = gamma[i]; bet[i] = beta[i]; b2l[i] = b2[i]; }
  if (tid < HDIM) b1l[tid] = b1[tid];

  // ---- zero hid tile (covers K-padding 48..63 for GEMM2 A fragments) ----
  {
    float* hz = (float*)hlds;
    for (int i = lane; i < (int)(16 * HID_STRIDE) / 2; i += 32) hz[i] = 0.f;
  }

  // ---- stage weight fragment pool into LDS ----
  if (use_ws) {
    // fast path: async DMA memory->LDS (GLOBAL_LOAD_ASYNC_TO_LDS_B128, ASYNCcnt),
    // no VGPR round-trip; fragments are already in final layout, copy is contiguous.
    const unsigned lds_base = (unsigned)(uintptr_t)smem;   // flat LDS addr low 32b == LDS offset
    for (int i = tid; i < (int)(WF_BYTES / 16u); i += 128) {
      unsigned ldsa = lds_base + (unsigned)i * 16u;
      unsigned goff = (unsigned)i * 16u;
      asm volatile("global_load_async_to_lds_b128 %0, %1, %2"
                   :: "v"(ldsa), "v"(goff), "s"(wsrc) : "memory");
    }
    asm volatile("s_wait_asynccnt 0x0" ::: "memory");
  } else {
    // fallback: swizzle fp32 weights in-kernel
    for (int u = tid; u < 72 * 256; u += 128) stage_one_w1(u, W1, smem + OFF_W1F);
    for (int u = tid; u < 96 * 256; u += 128) stage_one_w2(u, W2, smem + OFF_W2F);
  }

  __syncthreads();

  const long t0 = ((long)blockIdx.x * WAVES + wave) * 16;

  // ---- LayerNorm (fp32) -> f16 y rows in LDS; 4 rows of loads in flight ----
  for (int rb = 0; rb < 16; rb += 4) {
    float4 v[4][6];
#pragma unroll
    for (int rr = 0; rr < 4; ++rr) {
      const float4* xp = (const float4*)(x + (t0 + rb + rr) * CDIM);
#pragma unroll
      for (int i = 0; i < 6; ++i) v[rr][i] = xp[lane + 32 * i];
    }
#pragma unroll
    for (int rr = 0; rr < 4; ++rr) {
      float s = 0.f, ss = 0.f;
#pragma unroll
      for (int i = 0; i < 6; ++i) {
        s  += v[rr][i].x + v[rr][i].y + v[rr][i].z + v[rr][i].w;
        ss += v[rr][i].x * v[rr][i].x + v[rr][i].y * v[rr][i].y
            + v[rr][i].z * v[rr][i].z + v[rr][i].w * v[rr][i].w;
      }
#pragma unroll
      for (int m = 16; m >= 1; m >>= 1) { s += __shfl_xor(s, m); ss += __shfl_xor(ss, m); }
      const float mean = s * (1.f / 768.f);
      const float var  = ss * (1.f / 768.f) - mean * mean;
      const float inv  = rsqrtf(var + 1e-5f);
#pragma unroll
      for (int i = 0; i < 6; ++i) {
        const int c0 = (lane + 32 * i) * 4;
        float4 gv = *(const float4*)(gam + c0);
        float4 bv = *(const float4*)(bet + c0);
        v4h p;
        p[0] = (_Float16)((v[rr][i].x - mean) * inv * gv.x + bv.x);
        p[1] = (_Float16)((v[rr][i].y - mean) * inv * gv.y + bv.y);
        p[2] = (_Float16)((v[rr][i].z - mean) * inv * gv.z + bv.z);
        p[3] = (_Float16)((v[rr][i].w - mean) * inv * gv.w + bv.w);
        *(v4h*)(ylds + (rb + rr) * Y_STRIDE + c0) = p;
      }
    }
  }

  // keep this tile's x rows warm for the residual re-read (global_prefetch_b8)
#pragma unroll
  for (int i = 0; i < 12; ++i)
    __builtin_prefetch((const void*)(x + t0 * CDIM + (lane + 32 * i) * 32), 0, 1);

  // ---- GEMM1: hid[16x48] = y[16x768] @ W1, f16 WMMA with f32 accumulate ----
  // A k order used here (contiguous, same as B): lane<16 -> M=lane, K=k0..k0+15;
  //                          lane>=16 -> M=lane-16, K=k0+16..k0+31.
  const int rowN  = lane & 15;
  const int selK  = (lane >> 4) << 4;   // 0 or 16 (f16 elems)
  const int Mbase = (lane >> 4) << 3;

  v8f acc[3] = {};
  for (int kc = 0; kc < 24; ++kc) {
    const _Float16* ap = ylds + rowN * Y_STRIDE + kc * 32 + selK;
    v16h a = cat8(*(const v8h*)ap, *(const v8h*)(ap + 8));
#pragma unroll
    for (int nt = 0; nt < 3; ++nt) {
      const char* fb = smem + OFF_W1F + (unsigned)(kc * 3 + nt) * 1024u;
      v16h b = cat8(*(const v8h*)(fb + lane * 16), *(const v8h*)(fb + 512 + lane * 16));
      acc[nt] = __builtin_amdgcn_wmma_f32_16x16x32_f16(
          false, a, false, b, (short)0, acc[nt], false, false);
    }
  }

  // ---- bias + QuickGELU on accumulators, spill hid tile to LDS as f16 ----
  // C/D layout: lane<16 -> N=lane, M=v; lane>=16 -> N=lane-16, M=v+8.
#pragma unroll
  for (int nt = 0; nt < 3; ++nt) {
    const float bias = b1l[nt * 16 + rowN];
#pragma unroll
    for (int vv = 0; vv < 8; ++vv) {
      float h = acc[nt][vv] + bias;
      float g = h / (1.f + __expf(-1.702f * h));
      hlds[(Mbase + vv) * HID_STRIDE + nt * 16 + rowN] = (_Float16)g;
    }
  }

  // ---- GEMM2: out[16x768] = hid[16x64(padded)] @ W2 ----
  const _Float16* hp = hlds + rowN * HID_STRIDE + selK;
  v16h ha0 = cat8(*(const v8h*)hp,        *(const v8h*)(hp + 8));
  v16h ha1 = cat8(*(const v8h*)(hp + 32), *(const v8h*)(hp + 40));

  for (int nt = 0; nt < 48; ++nt) {
    const char* fb = smem + OFF_W2F + (unsigned)(nt * 2) * 1024u;
    v16h bf0 = cat8(*(const v8h*)(fb + lane * 16),        *(const v8h*)(fb + 512  + lane * 16));
    v16h bf1 = cat8(*(const v8h*)(fb + 1024 + lane * 16), *(const v8h*)(fb + 1536 + lane * 16));
    v8f oacc = {};
    oacc = __builtin_amdgcn_wmma_f32_16x16x32_f16(false, ha0, false, bf0, (short)0, oacc, false, false);
    oacc = __builtin_amdgcn_wmma_f32_16x16x32_f16(false, ha1, false, bf1, (short)0, oacc, false, false);
    const int  cc   = nt * 16 + rowN;
    const float bias = b2l[cc];
#pragma unroll
    for (int vv = 0; vv < 8; ++vv) {
      long idx = (t0 + Mbase + vv) * CDIM + cc;
      float res = oacc[vv] + bias + x[idx];   // residual re-read: L2-resident
      __builtin_nontemporal_store(res, out + idx);
    }
  }
}

extern "C" void kernel_launch(void* const* d_in, const int* in_sizes, int n_in,
                              void* d_out, int out_size, void* d_ws, size_t ws_size,
                              hipStream_t stream) {
  (void)n_in; (void)out_size;
  const float* x     = (const float*)d_in[0];
  const float* W1    = (const float*)d_in[1];
  const float* b1    = (const float*)d_in[2];
  const float* W2    = (const float*)d_in[3];
  const float* b2    = (const float*)d_in[4];
  const float* gamma = (const float*)d_in[5];
  const float* beta  = (const float*)d_in[6];
  float* out = (float*)d_out;

  const int tokens = in_sizes[0] / CDIM;       // 131072
  const int wgs    = tokens / (WAVES * 16);    // 2048

  const int use_ws = (ws_size >= (size_t)WF_BYTES) ? 1 : 0;
  if (use_ws) {
    prep_weights_kernel<<<(72 * 256 + 96 * 256) / 256, 256, 0, stream>>>(W1, W2, (char*)d_ws);
  }

  hipFuncSetAttribute(reinterpret_cast<const void*>(fused_ln_mlp_kernel),
                      hipFuncAttributeMaxDynamicSharedMemorySize, (int)LDS_TOTAL);
  fused_ln_mlp_kernel<<<wgs, 128, LDS_TOTAL, stream>>>(
      x, W1, b1, W2, b2, gamma, beta, out, (const char*)d_ws, use_ws);
}
